// Gen_64699387347198
// MI455X (gfx1250) — matrix operand order repaired and checked by reading, not compiled, not run on hardware
//
#include <hip/hip_runtime.h>

// ---------------------------------------------------------------------------
// Types for CDNA5 WMMA fragments (wave32)
// ---------------------------------------------------------------------------
typedef _Float16 v16h __attribute__((ext_vector_type(16)));   // A/B frag: 8 VGPRs
typedef _Float16 h4   __attribute__((ext_vector_type(4)));
typedef float    v8f  __attribute__((ext_vector_type(8)));    // C/D frag: 8 VGPRs
typedef unsigned int u32x4 __attribute__((ext_vector_type(4)));

#define N_ROWS   65536
#define DIM      256
#define H1       512
#define H2       2048
#define MTILE    128          // rows per workgroup in fused kernel
#define NBLOCKS_MAIN (N_ROWS / MTILE)

__device__ __forceinline__ float fast_tanh(float x) {
#if __has_builtin(__builtin_amdgcn_tanhf)
  return __builtin_amdgcn_tanhf(x);
#else
  return tanhf(x);
#endif
}

// A-fragment (16x32 f16, MxK): lane = r + 16h, r=M. VGPR0-3: K=k0+8h+{0..7},
// VGPR4-7: K=k0+16+8h+{0..7}.  p points at (row r of a row-major [*,256] src)+k0.
__device__ __forceinline__ v16h load_fragA(const _Float16* p, int h) {
  union { v16h v; u32x4 q[2]; } u;
  u.q[0] = *(const u32x4*)(p + 8 * h);
  u.q[1] = *(const u32x4*)(p + 16 + 8 * h);
  return u.v;
}

// B-fragment (32x16 f16, KxN) loaded from row-major W[N,K] (i.e. x @ W^T):
// lane = n + 16h, VGPR0-3: K=k0+8h+{0..7}, VGPR4-7: K=k0+16+8h+{0..7}
// (same as A).  p points at (row n of row-major [*,256]) + k0.
__device__ __forceinline__ v16h load_fragB(const _Float16* p, int h) {
  union { v16h v; u32x4 q[2]; } u;
  u.q[0] = *(const u32x4*)(p + 8 * h);
  u.q[1] = *(const u32x4*)(p + 16 + 8 * h);
  return u.v;
}

// ---------------------------------------------------------------------------
// 1) BN column partial sums (deterministic two-stage; no float atomics)
// ---------------------------------------------------------------------------
__global__ void k_bn_reduce(const float* __restrict__ x,
                            float* __restrict__ psum, float* __restrict__ psumsq) {
  const int c = threadIdx.x;          // column
  const int b = blockIdx.x;           // 256-row slab
  const float* xp = x + (size_t)b * 256 * DIM + c;
  float s = 0.f, s2 = 0.f;
  for (int r = 0; r < 256; ++r) {
    float v = xp[(size_t)r * DIM];
    s += v;
    s2 = fmaf(v, v, s2);
  }
  psum[b * DIM + c]   = s;
  psumsq[b * DIM + c] = s2;
}

__global__ void k_bn_finalize(const float* __restrict__ psum,
                              const float* __restrict__ psumsq,
                              const float* __restrict__ gamma,
                              const float* __restrict__ beta,
                              float* __restrict__ scale, float* __restrict__ shift) {
  const int c = threadIdx.x;
  float s = 0.f, s2 = 0.f;
  for (int b = 0; b < 256; ++b) {
    s  += psum[b * DIM + c];
    s2 += psumsq[b * DIM + c];
  }
  const float inv_n = 1.0f / (float)N_ROWS;
  float mean = s * inv_n;
  float var  = s2 * inv_n - mean * mean;
  float sc   = gamma[c] * rsqrtf(var + 1e-5f);
  scale[c] = sc;
  shift[c] = beta[c] - mean * sc;
}

// ---------------------------------------------------------------------------
// 2) Weight collapse: T1 = Wf @ W2  (256x512), fp32 tiled GEMM
// ---------------------------------------------------------------------------
__global__ void k_t1(const float* __restrict__ Wf, const float* __restrict__ W2,
                     float* __restrict__ T1) {
  __shared__ float sA[16][17];
  __shared__ float sB[16][17];
  const int tx = threadIdx.x, ty = threadIdx.y;
  const int row = blockIdx.y * 16 + ty;   // < 256
  const int col = blockIdx.x * 16 + tx;   // < 512
  float acc = 0.f;
  for (int kt = 0; kt < H2 / 16; ++kt) {
    sA[ty][tx] = Wf[(size_t)row * H2 + kt * 16 + tx];
    sB[ty][tx] = W2[(size_t)(kt * 16 + ty) * H1 + col];
    __syncthreads();
#pragma unroll
    for (int i = 0; i < 16; ++i) acc = fmaf(sA[ty][i], sB[i][tx], acc);
    __syncthreads();
  }
  T1[(size_t)row * H1 + col] = acc;
}

// Weff = T1 @ W1  (256x256), stored f16 row-major [N,K]
__global__ void k_weff(const float* __restrict__ T1, const float* __restrict__ W1,
                       _Float16* __restrict__ weffh) {
  __shared__ float sA[16][17];
  __shared__ float sB[16][17];
  const int tx = threadIdx.x, ty = threadIdx.y;
  const int row = blockIdx.y * 16 + ty;   // < 256 (output dim N)
  const int col = blockIdx.x * 16 + tx;   // < 256 (input dim K)
  float acc = 0.f;
  for (int kt = 0; kt < H1 / 16; ++kt) {
    sA[ty][tx] = T1[(size_t)row * H1 + kt * 16 + tx];
    sB[ty][tx] = W1[(size_t)(kt * 16 + ty) * DIM + col];
    __syncthreads();
#pragma unroll
    for (int i = 0; i < 16; ++i) acc = fmaf(sA[ty][i], sB[i][tx], acc);
    __syncthreads();
  }
  weffh[(size_t)row * DIM + col] = (_Float16)acc;
}

// tvec = W2 @ b1 + b2   (2048)
__global__ void k_tvec(const float* __restrict__ W2, const float* __restrict__ b1,
                       const float* __restrict__ b2, float* __restrict__ tvec) {
  const int i = blockIdx.x * blockDim.x + threadIdx.x;   // < 2048
  float acc = b2[i];
  const float* w = W2 + (size_t)i * H1;
  for (int k = 0; k < H1; ++k) acc = fmaf(w[k], b1[k], acc);
  tvec[i] = acc;
}

// beff = Wf @ tvec + bf   (256)
__global__ void k_beff(const float* __restrict__ Wf, const float* __restrict__ tvec,
                       const float* __restrict__ bf, float* __restrict__ beff) {
  const int j = threadIdx.x;   // < 256
  float acc = bf[j];
  const float* w = Wf + (size_t)j * H2;
  for (int k = 0; k < H2; ++k) acc = fmaf(w[k], tvec[k], acc);
  beff[j] = acc;
}

// ---------------------------------------------------------------------------
// 3) Fused main pass: xn = x*scale+shift ; raw = tanh(xn) ;
//    out = tanh(xn @ Weff^T + beff)   — GEMM on v_wmma_f32_16x16x32_f16
//    8 wave32s per block, 128-row tile, each wave owns 32 output columns.
// ---------------------------------------------------------------------------
__global__ __launch_bounds__(256) void k_fused(
    const float* __restrict__ x,
    const float* __restrict__ scale, const float* __restrict__ shift,
    const _Float16* __restrict__ weffh, const float* __restrict__ beff,
    float* __restrict__ out, float* __restrict__ raw) {
  __shared__ _Float16 xn_lds[MTILE * DIM];   // 64 KB

  const int t = threadIdx.x;
  const size_t rowbase = (size_t)blockIdx.x * MTILE;

  // ---- Phase A: coalesced load x, BN transform, emit raw_feats, xn->f16 LDS
  const float4* x4   = (const float4*)x;
  float4*       raw4 = (float4*)raw;
  const float4* sc4  = (const float4*)scale;
  const float4* sh4  = (const float4*)shift;
#pragma unroll 4
  for (int it = 0; it < 32; ++it) {
    const int idx4 = it * 256 + t;        // [0, 8192)
    const int row  = idx4 >> 6;           // 64 float4 per row
    const int col4 = idx4 & 63;
    float4 v = x4[(rowbase + row) * 64 + col4];
    float4 s = sc4[col4];
    float4 b = sh4[col4];
    float4 xn;
    xn.x = fmaf(v.x, s.x, b.x);
    xn.y = fmaf(v.y, s.y, b.y);
    xn.z = fmaf(v.z, s.z, b.z);
    xn.w = fmaf(v.w, s.w, b.w);
    float4 rt;
    rt.x = fast_tanh(xn.x);
    rt.y = fast_tanh(xn.y);
    rt.z = fast_tanh(xn.z);
    rt.w = fast_tanh(xn.w);
    raw4[(rowbase + row) * 64 + col4] = rt;
    h4 hv;
    hv.x = (_Float16)xn.x;
    hv.y = (_Float16)xn.y;
    hv.z = (_Float16)xn.z;
    hv.w = (_Float16)xn.w;
    *(h4*)&xn_lds[row * DIM + col4 * 4] = hv;
  }
  __syncthreads();

  // ---- Phase B: WMMA GEMM. wave w -> output columns [32w, 32w+32)
  const int wave = t >> 5;
  const int lane = t & 31;
  const int r = lane & 15;
  const int h = lane >> 4;
  const int n0 = wave * 32;

  v8f acc[8][2];
#pragma unroll
  for (int mt = 0; mt < 8; ++mt) {
    acc[mt][0] = {};
    acc[mt][1] = {};
  }

  for (int kk = 0; kk < 8; ++kk) {
    const int k0 = kk * 32;
    v16h b0 = load_fragB(weffh + (size_t)(n0 + r) * DIM + k0, h);
    v16h b1 = load_fragB(weffh + (size_t)(n0 + 16 + r) * DIM + k0, h);
#pragma unroll
    for (int mt = 0; mt < 8; ++mt) {
      v16h a = load_fragA(&xn_lds[(mt * 16 + r) * DIM + k0], h);
      acc[mt][0] = __builtin_amdgcn_wmma_f32_16x16x32_f16(
          false, a, false, b0, (short)0, acc[mt][0], false, false);
      acc[mt][1] = __builtin_amdgcn_wmma_f32_16x16x32_f16(
          false, a, false, b1, (short)0, acc[mt][1], false, false);
    }
  }

  // ---- Epilogue: + beff, tanh, store.  D layout: n = n0 + nt*16 + r,
  // m = mt*16 + v + 8h  (v = accumulator VGPR index)
  const float bA = beff[n0 + r];
  const float bB = beff[n0 + 16 + r];
#pragma unroll
  for (int mt = 0; mt < 8; ++mt) {
#pragma unroll
    for (int v = 0; v < 8; ++v) {
      const size_t o = (rowbase + mt * 16 + v + 8 * h) * DIM;
      out[o + n0 + r]      = fast_tanh(acc[mt][0][v] + bA);
      out[o + n0 + 16 + r] = fast_tanh(acc[mt][1][v] + bB);
    }
  }
}

// ---------------------------------------------------------------------------
// Host launcher
// ---------------------------------------------------------------------------
extern "C" void kernel_launch(void* const* d_in, const int* in_sizes, int n_in,
                              void* d_out, int out_size, void* d_ws, size_t ws_size,
                              hipStream_t stream) {
  const float* x     = (const float*)d_in[0];
  const float* gamma = (const float*)d_in[1];
  const float* beta  = (const float*)d_in[2];
  const float* W1    = (const float*)d_in[3];
  const float* b1    = (const float*)d_in[4];
  const float* W2    = (const float*)d_in[5];
  const float* b2    = (const float*)d_in[6];
  const float* Wf    = (const float*)d_in[7];
  const float* bf    = (const float*)d_in[8];

  float* out = (float*)d_out;
  float* raw = out + (size_t)N_ROWS * DIM;

  // workspace layout (fp32 unless noted)
  float* wsf    = (float*)d_ws;
  float* psum   = wsf;                    // 65536
  float* psumsq = psum + 65536;           // 65536
  float* scale  = psumsq + 65536;         // 256
  float* shift  = scale + 256;            // 256
  float* tvec   = shift + 256;            // 2048
  float* beff   = tvec + 2048;            // 256
  float* T1     = beff + 256;             // 256*512
  _Float16* weffh = (_Float16*)(T1 + 256 * 512);   // 256*256 f16 (16B aligned)

  // BN statistics (deterministic two-stage)
  k_bn_reduce<<<256, 256, 0, stream>>>(x, psum, psumsq);
  k_bn_finalize<<<1, 256, 0, stream>>>(psum, psumsq, gamma, beta, scale, shift);

  // Collapse the linear stack: Weff = Wf@W2@W1, beff = Wf@(W2@b1+b2)+bf
  k_t1<<<dim3(H1 / 16, DIM / 16), dim3(16, 16), 0, stream>>>(Wf, W2, T1);
  k_weff<<<dim3(DIM / 16, DIM / 16), dim3(16, 16), 0, stream>>>(T1, W1, weffh);
  k_tvec<<<H2 / 256, 256, 0, stream>>>(W2, b1, b2, tvec);
  k_beff<<<1, 256, 0, stream>>>(Wf, tvec, bf, beff);

  // Fused BN + tanh(raw) + WMMA GEMM + bias + tanh(out)
  k_fused<<<NBLOCKS_MAIN, 256, 0, stream>>>(x, scale, shift, weffh, beff, out, raw);
}
